// TAGNet_73100343378448
// MI455X (gfx1250) — compile-verified, matrix operand order checked by reading, not run on hardware
//
#include <hip/hip_runtime.h>
#include <math.h>

#define Bsz 64
#define Lseq 512
#define Cdim 256
#define Hh 4
#define Dd 64
#define BL (Bsz * Lseq)
#define NEGBIG -1e9f

typedef __attribute__((ext_vector_type(16))) __bf16 v16bf;
typedef __attribute__((ext_vector_type(8)))  float  v8f;

union FragBF { v16bf v; unsigned int u[8]; };

__device__ __forceinline__ unsigned short f2bf(float f) {
    unsigned int u = __builtin_bit_cast(unsigned int, f);
    unsigned int r = u + 0x7FFFu + ((u >> 16) & 1u);   // round-to-nearest-even
    return (unsigned short)(r >> 16);
}

// ---------------------------------------------------------------------------
// GEMM: C[M,N] = A[M,K] @ W[N,K]^T (+bias, optional relu), bf16 WMMA, f32 acc
// block = 256 thr (8 waves). Block tile 128x128, wave tile 32x64 (2x4 WMMA).
// M%128==0, N%128==0, K%32==0 by construction.
// ---------------------------------------------------------------------------
#define TM 128
#define TN 128
#define TK 32
#define ASTR 34   // padded bf16 row stride in LDS

template<int HASBIAS, int RELU>
__global__ __launch_bounds__(256) void gemm_bf16_wmma(
    const float* __restrict__ A, const float* __restrict__ W,
    const float* __restrict__ bias, float* __restrict__ C,
    int M, int N, int K)
{
    __shared__ __align__(16) unsigned short lA[TM * ASTR];
    __shared__ __align__(16) unsigned short lB[TN * ASTR];

    const int tid  = threadIdx.x;
    const int m0   = blockIdx.x * TM;
    const int n0   = blockIdx.y * TN;
    const int wave = tid >> 5;
    const int lane = tid & 31;
    const int waveM = wave >> 1;      // 0..3 -> 32-row slice
    const int waveN = wave & 1;       // 0..1 -> 64-col slice
    const int half  = lane >> 4;
    const int l16   = lane & 15;

    v8f acc[2][4];
    const v8f zf = {0.f,0.f,0.f,0.f,0.f,0.f,0.f,0.f};
    #pragma unroll
    for (int ti = 0; ti < 2; ++ti)
        #pragma unroll
        for (int tj = 0; tj < 4; ++tj) acc[ti][tj] = zf;

    for (int k0 = 0; k0 < K; k0 += TK) {
        __syncthreads();
        // 128x32 f32 -> bf16 LDS, vectorized float4 (4 vec loads / thread)
        #pragma unroll
        for (int idx = tid; idx < TM * TK / 4; idx += 256) {
            int r = idx >> 3, c4 = (idx & 7) * 4;
            float4 v = *(const float4*)&A[(size_t)(m0 + r) * K + k0 + c4];
            int base = r * ASTR + c4;
            lA[base + 0] = f2bf(v.x); lA[base + 1] = f2bf(v.y);
            lA[base + 2] = f2bf(v.z); lA[base + 3] = f2bf(v.w);
        }
        #pragma unroll
        for (int idx = tid; idx < TN * TK / 4; idx += 256) {
            int r = idx >> 3, c4 = (idx & 7) * 4;
            float4 v = *(const float4*)&W[(size_t)(n0 + r) * K + k0 + c4];
            int base = r * ASTR + c4;
            lB[base + 0] = f2bf(v.x); lB[base + 1] = f2bf(v.y);
            lB[base + 2] = f2bf(v.z); lB[base + 3] = f2bf(v.w);
        }
        __syncthreads();

        FragBF a[2], b[4];
        #pragma unroll
        for (int t = 0; t < 2; ++t) {
            int rowA = waveM * 32 + t * 16 + l16;
            #pragma unroll
            for (int v = 0; v < 8; ++v) {
                int kk = (v >> 2) * 16 + half * 8 + (v & 3) * 2;
                a[t].u[v] = *(const unsigned int*)&lA[rowA * ASTR + kk];
            }
        }
        #pragma unroll
        for (int t = 0; t < 4; ++t) {
            int rowB = waveN * 64 + t * 16 + l16;
            #pragma unroll
            for (int v = 0; v < 8; ++v) {
                int kk = (v >> 2) * 16 + half * 8 + (v & 3) * 2;
                b[t].u[v] = *(const unsigned int*)&lB[rowB * ASTR + kk];
            }
        }
        #pragma unroll
        for (int ti = 0; ti < 2; ++ti)
            #pragma unroll
            for (int tj = 0; tj < 4; ++tj)
                acc[ti][tj] = __builtin_amdgcn_wmma_f32_16x16x32_bf16(
                    false, a[ti].v, false, b[tj].v,
                    (short)0, acc[ti][tj], false, false);
    }

    #pragma unroll
    for (int ti = 0; ti < 2; ++ti)
        #pragma unroll
        for (int tj = 0; tj < 4; ++tj) {
            int row = m0 + waveM * 32 + ti * 16 + 8 * half;
            int col = n0 + waveN * 64 + tj * 16 + l16;
            float bv = HASBIAS ? bias[col] : 0.f;
            float* cp = &C[(size_t)row * N + col];
            #pragma unroll
            for (int v = 0; v < 8; ++v) {
                float d = acc[ti][tj][v] + bv;
                if (RELU) d = fmaxf(d, 0.f);
                cp[(size_t)v * N] = d;
            }
        }
}

static inline void launch_gemm(const float* A, const float* W, const float* bias,
                               float* C, int M, int N, int K, int relu,
                               hipStream_t s) {
    dim3 g(M / TM, N / TN);
    if (bias) {
        if (relu) gemm_bf16_wmma<1,1><<<g, 256, 0, s>>>(A, W, bias, C, M, N, K);
        else      gemm_bf16_wmma<1,0><<<g, 256, 0, s>>>(A, W, bias, C, M, N, K);
    } else {
        gemm_bf16_wmma<0,0><<<g, 256, 0, s>>>(A, W, bias, C, M, N, K);
    }
}

// ---------------------------------------------------------------------------
// h = (emb[x] + pos*mf)*mf ; mask
// ---------------------------------------------------------------------------
__global__ __launch_bounds__(256) void embed_kernel(
    const int* __restrict__ x, const float* __restrict__ emb,
    const float* __restrict__ pos, float* __restrict__ h, int* __restrict__ maskA)
{
    int bl = blockIdx.x, c = threadIdx.x;
    int l = bl & (Lseq - 1);
    int tok = x[bl];
    float mf = (tok != 0) ? 1.f : 0.f;
    h[(size_t)bl * Cdim + c] = (emb[(size_t)tok * Cdim + c] + pos[l * Cdim + c] * mf) * mf;
    if (c == 0) maskA[bl] = (tok != 0);
}

// ---------------------------------------------------------------------------
// per-batch scan: logical = clip(cumsum(mask)-1,0), inv[n] = raw idx of n-th
// valid token, nvalid
// ---------------------------------------------------------------------------
__global__ void prep_kernel(const int* __restrict__ maskA,
                            int* __restrict__ logical, int* __restrict__ inv,
                            int* __restrict__ nvalid)
{
    if (threadIdx.x != 0) return;
    int b = blockIdx.x, cnt = 0;
    for (int i = 0; i < Lseq; ++i) {
        if (maskA[b * Lseq + i]) { inv[b * Lseq + cnt] = i; cnt++; }
        logical[b * Lseq + i] = (cnt - 1 > 0) ? (cnt - 1) : 0;
    }
    nvalid[b] = cnt;
}

// ---------------------------------------------------------------------------
// GAT attention scores: s/d = <proj_head, a_vec> for past and future branches
// block per (b,l): 4 heads x 64 lanes
// ---------------------------------------------------------------------------
__global__ __launch_bounds__(256) void score_kernel(
    const float* __restrict__ projP, const float* __restrict__ projF,
    const float* __restrict__ asP, const float* __restrict__ adP,
    const float* __restrict__ asF, const float* __restrict__ adF,
    float* __restrict__ sp, float* __restrict__ dp,
    float* __restrict__ sf, float* __restrict__ df)
{
    __shared__ float red[4][256];
    int bl = blockIdx.x, tid = threadIdx.x;
    int head = tid >> 6, dd = tid & 63, ch = head * 64 + dd;
    float vp = projP[(size_t)bl * Cdim + ch];
    float vf = projF[(size_t)bl * Cdim + ch];
    red[0][tid] = vp * asP[ch];
    red[1][tid] = vp * adP[ch];
    red[2][tid] = vf * asF[ch];
    red[3][tid] = vf * adF[ch];
    __syncthreads();
    for (int s = 32; s > 0; s >>= 1) {
        if (dd < s)
            #pragma unroll
            for (int q = 0; q < 4; ++q) red[q][tid] += red[q][tid + s];
        __syncthreads();
    }
    if (dd == 0) {
        int b = bl >> 9, i = bl & (Lseq - 1);
        int o = (b * Hh + head) * Lseq + i;
        sp[o] = red[0][tid]; dp[o] = red[1][tid];
        sf[o] = red[2][tid]; df[o] = red[3][tid];
    }
}

// ---------------------------------------------------------------------------
// sparse banded GAT: per (b,i), <=3 dilated neighbors + self; writes mcat
// (past -> cols [0,256), future -> cols [256,512))
// ---------------------------------------------------------------------------
__global__ __launch_bounds__(256) void gat_kernel(
    const float* __restrict__ projP, const float* __restrict__ projF,
    const float* __restrict__ sp, const float* __restrict__ dp,
    const float* __restrict__ sf, const float* __restrict__ df,
    const int* __restrict__ logical, const int* __restrict__ inv,
    const int* __restrict__ nvalid, const int* __restrict__ maskA,
    float* __restrict__ mcat, int dil)
{
    int bl = blockIdx.x;
    int b = bl >> 9, i = bl & (Lseq - 1);
    int tid = threadIdx.x, head = tid >> 6, dd = tid & 63, ch = head * 64 + dd;
    int n = logical[bl], mi = maskA[bl], nv = nvalid[b];
    int sbase = (b * Hh + head) * Lseq;

    // ---- "past" branch (adj_p: logical[j] = n + k*dil) ----
    {
        float si = sp[sbase + i];
        int js[3]; float ev[4];
        float e0 = si + dp[sbase + i];
        ev[0] = e0 > 0.f ? e0 : 0.2f * e0;
        #pragma unroll
        for (int k = 1; k <= 3; ++k) {
            int tn = n + k * dil, j = -1;
            if (mi && tn < nv) j = inv[b * Lseq + tn];
            js[k - 1] = j;
            float e = NEGBIG;
            if (j >= 0) { float t = si + dp[sbase + j]; e = t > 0.f ? t : 0.2f * t; }
            ev[k] = e;
        }
        float mx = fmaxf(fmaxf(ev[0], ev[1]), fmaxf(ev[2], ev[3]));
        float w[4], den = 0.f;
        #pragma unroll
        for (int k = 0; k < 4; ++k) { w[k] = __expf(ev[k] - mx); den += w[k]; }
        float out = w[0] * projP[(size_t)bl * Cdim + ch];
        #pragma unroll
        for (int k = 1; k <= 3; ++k)
            if (js[k - 1] >= 0)
                out += w[k] * projP[(size_t)(b * Lseq + js[k - 1]) * Cdim + ch];
        mcat[(size_t)bl * (2 * Cdim) + ch] = out / den;
    }
    // ---- "future" branch (adj_f: logical[j] = n - k*dil) ----
    {
        float si = sf[sbase + i];
        int js[3]; float ev[4];
        float e0 = si + df[sbase + i];
        ev[0] = e0 > 0.f ? e0 : 0.2f * e0;
        #pragma unroll
        for (int k = 1; k <= 3; ++k) {
            int tn = n - k * dil, j = -1;
            if (mi && tn >= 0) j = inv[b * Lseq + tn];
            js[k - 1] = j;
            float e = NEGBIG;
            if (j >= 0) { float t = si + df[sbase + j]; e = t > 0.f ? t : 0.2f * t; }
            ev[k] = e;
        }
        float mx = fmaxf(fmaxf(ev[0], ev[1]), fmaxf(ev[2], ev[3]));
        float w[4], den = 0.f;
        #pragma unroll
        for (int k = 0; k < 4; ++k) { w[k] = __expf(ev[k] - mx); den += w[k]; }
        float out = w[0] * projF[(size_t)bl * Cdim + ch];
        #pragma unroll
        for (int k = 1; k <= 3; ++k)
            if (js[k - 1] >= 0)
                out += w[k] * projF[(size_t)(b * Lseq + js[k - 1]) * Cdim + ch];
        mcat[(size_t)bl * (2 * Cdim) + Cdim + ch] = out / den;
    }
}

// ---------------------------------------------------------------------------
// GRU gates + LayerNorm + mask; h updated in place. block per (b,l).
// ---------------------------------------------------------------------------
__global__ __launch_bounds__(256) void gru_ln_kernel(
    const float* __restrict__ gi, const float* __restrict__ gh,
    const float* __restrict__ lng, const float* __restrict__ lnb,
    const int* __restrict__ maskA, float* __restrict__ h)
{
    __shared__ float s1[256], s2[256];
    int bl = blockIdx.x, c = threadIdx.x;
    size_t g = (size_t)bl * (3 * Cdim);
    float ir = gi[g + c], iz = gi[g + Cdim + c], in_ = gi[g + 2 * Cdim + c];
    float hr = gh[g + c], hz = gh[g + Cdim + c], hn  = gh[g + 2 * Cdim + c];
    float hv = h[(size_t)bl * Cdim + c];
    float r = 1.f / (1.f + __expf(-(ir + hr)));
    float z = 1.f / (1.f + __expf(-(iz + hz)));
    float nn = tanhf(in_ + r * hn);
    float x = (1.f - z) * nn + z * hv;
    s1[c] = x; s2[c] = x * x;
    __syncthreads();
    for (int s = 128; s > 0; s >>= 1) {
        if (c < s) { s1[c] += s1[c + s]; s2[c] += s2[c + s]; }
        __syncthreads();
    }
    float mu = s1[0] * (1.f / Cdim);
    float var = s2[0] * (1.f / Cdim) - mu * mu;
    float y = (x - mu) * rsqrtf(var + 1e-5f) * lng[c] + lnb[c];
    h[(size_t)bl * Cdim + c] = maskA[bl] ? y : 0.f;
}

// ---------------------------------------------------------------------------
// masked attention pooling + two linear heads. block per batch element.
// ---------------------------------------------------------------------------
__global__ __launch_bounds__(256) void pool_head_kernel(
    const float* __restrict__ h, const int* __restrict__ maskA,
    const float* __restrict__ pw, const float* __restrict__ pb,
    const float* __restrict__ h0W, const float* __restrict__ h0b,
    const float* __restrict__ h1W, const float* __restrict__ h1b,
    float* __restrict__ out)
{
    __shared__ float sc[Lseq];
    __shared__ float red[256];
    __shared__ float pooled[Cdim];
    int b = blockIdx.x, tid = threadIdx.x;

    for (int l = tid; l < Lseq; l += 256) {
        const float* hp = h + (size_t)(b * Lseq + l) * Cdim;
        float d = 0.f;
        for (int c = 0; c < Cdim; ++c) d += hp[c] * pw[c];
        sc[l] = maskA[b * Lseq + l] ? (d + pb[0]) : NEGBIG;
    }
    __syncthreads();
    red[tid] = fmaxf(sc[tid], sc[tid + 256]);
    __syncthreads();
    for (int s = 128; s > 0; s >>= 1) {
        if (tid < s) red[tid] = fmaxf(red[tid], red[tid + s]);
        __syncthreads();
    }
    float mx = red[0];
    __syncthreads();
    float part = 0.f;
    for (int l = tid; l < Lseq; l += 256) {
        float e = __expf(sc[l] - mx); sc[l] = e; part += e;
    }
    red[tid] = part;
    __syncthreads();
    for (int s = 128; s > 0; s >>= 1) {
        if (tid < s) red[tid] += red[tid + s];
        __syncthreads();
    }
    float invden = 1.f / red[0];

    float acc = 0.f;
    for (int l = 0; l < Lseq; ++l)
        acc += sc[l] * h[(size_t)(b * Lseq + l) * Cdim + tid];
    pooled[tid] = acc * invden;
    __syncthreads();

    if (tid < 50) {
        float d = h0b[tid];
        for (int c = 0; c < Cdim; ++c) d += pooled[c] * h0W[tid * Cdim + c];
        out[b * 70 + tid] = d;
    } else if (tid >= 64 && tid < 84) {
        int t = tid - 64;
        float d = h1b[t];
        for (int c = 0; c < Cdim; ++c) d += pooled[c] * h1W[t * Cdim + c];
        out[b * 70 + 50 + t] = d;
    }
}

// ---------------------------------------------------------------------------
extern "C" void kernel_launch(void* const* d_in, const int* in_sizes, int n_in,
                              void* d_out, int out_size, void* d_ws, size_t ws_size,
                              hipStream_t stream) {
    const int*   x     = (const int*)  d_in[0];
    const float* emb   = (const float*)d_in[1];
    const float* pos   = (const float*)d_in[2];
    const float* gpW   = (const float*)d_in[3];
    const float* gpAs  = (const float*)d_in[4];
    const float* gpAd  = (const float*)d_in[5];
    const float* gfW   = (const float*)d_in[6];
    const float* gfAs  = (const float*)d_in[7];
    const float* gfAd  = (const float*)d_in[8];
    const float* msgW  = (const float*)d_in[9];
    const float* msgB  = (const float*)d_in[10];
    const float* gWih  = (const float*)d_in[11];
    const float* gBih  = (const float*)d_in[12];
    const float* gWhh  = (const float*)d_in[13];
    const float* gBhh  = (const float*)d_in[14];
    const float* lnG   = (const float*)d_in[15];
    const float* lnB   = (const float*)d_in[16];
    const float* poolw = (const float*)d_in[17];
    const float* poolb = (const float*)d_in[18];
    const float* h0W   = (const float*)d_in[19];
    const float* h0b   = (const float*)d_in[20];
    const float* h1W   = (const float*)d_in[21];
    const float* h1b   = (const float*)d_in[22];
    float* out = (float*)d_out;

    char* ws = (char*)d_ws;
    size_t off = 0;
    auto alloc = [&](size_t bytes) -> void* {
        void* p = ws + off;
        off += (bytes + 255) & ~(size_t)255;
        return p;
    };
    float* h     = (float*)alloc((size_t)BL * Cdim * 4);
    float* projP = (float*)alloc((size_t)BL * Cdim * 4);
    float* projF = (float*)alloc((size_t)BL * Cdim * 4);
    float* mcat  = (float*)alloc((size_t)BL * 2 * Cdim * 4);
    float* msg   = (float*)alloc((size_t)BL * Cdim * 4);
    float* gi    = (float*)alloc((size_t)BL * 3 * Cdim * 4);
    float* gh    = (float*)alloc((size_t)BL * 3 * Cdim * 4);
    float* sp    = (float*)alloc((size_t)Bsz * Hh * Lseq * 4);
    float* dp    = (float*)alloc((size_t)Bsz * Hh * Lseq * 4);
    float* sf    = (float*)alloc((size_t)Bsz * Hh * Lseq * 4);
    float* df    = (float*)alloc((size_t)Bsz * Hh * Lseq * 4);
    int*   maskA = (int*)  alloc((size_t)BL * 4);
    int*   logic = (int*)  alloc((size_t)BL * 4);
    int*   invA  = (int*)  alloc((size_t)BL * 4);
    int*   nval  = (int*)  alloc((size_t)Bsz * 4);

    embed_kernel<<<BL, 256, 0, stream>>>(x, emb, pos, h, maskA);
    prep_kernel<<<Bsz, 32, 0, stream>>>(maskA, logic, invA, nval);

    const int DILS[3] = {1, 2, 4};
    for (int l = 0; l < 3; ++l) {
        // projections: [BL,256] @ [256,256]^T
        launch_gemm(h, gpW + (size_t)l * Cdim * Cdim, nullptr, projP,
                    BL, Cdim, Cdim, 0, stream);
        launch_gemm(h, gfW + (size_t)l * Cdim * Cdim, nullptr, projF,
                    BL, Cdim, Cdim, 0, stream);

        score_kernel<<<BL, 256, 0, stream>>>(
            projP, projF,
            gpAs + l * Hh * Dd, gpAd + l * Hh * Dd,
            gfAs + l * Hh * Dd, gfAd + l * Hh * Dd,
            sp, dp, sf, df);

        gat_kernel<<<BL, 256, 0, stream>>>(
            projP, projF, sp, dp, sf, df, logic, invA, nval, maskA, mcat, DILS[l]);

        // msg = relu(mcat @ msg_W^T + b): [BL,512] x [256,512]^T
        launch_gemm(mcat, msgW + (size_t)l * Cdim * 2 * Cdim, msgB + l * Cdim,
                    msg, BL, Cdim, 2 * Cdim, 1, stream);

        // GRU matmuls: [BL,256] x [768,256]^T
        launch_gemm(msg, gWih + (size_t)l * 3 * Cdim * Cdim, gBih + l * 3 * Cdim,
                    gi, BL, 3 * Cdim, Cdim, 0, stream);
        launch_gemm(h, gWhh + (size_t)l * 3 * Cdim * Cdim, gBhh + l * 3 * Cdim,
                    gh, BL, 3 * Cdim, Cdim, 0, stream);

        gru_ln_kernel<<<BL, 256, 0, stream>>>(
            gi, gh, lnG + l * Cdim, lnB + l * Cdim, maskA, h);
    }

    pool_head_kernel<<<Bsz, 256, 0, stream>>>(
        h, maskA, poolw, poolb, h0W, h0b, h1W, h1b, out);
}
